// OracleAD3D_46505905881350
// MI455X (gfx1250) — compile-verified, matrix operand order checked by reading, not run on hardware
//
#include <hip/hip_runtime.h>
#include <hip/hip_bf16.h>

typedef __attribute__((ext_vector_type(16))) _Float16 v16h;
typedef __attribute__((ext_vector_type(8)))  float    v8f;
typedef __attribute__((ext_vector_type(8)))  unsigned int v8u;

// ---------------------------------------------------------------------------
// Helpers: f16 WMMA fragment building (ISA 7.12.2 16-bit A layout).
// Lane l (0..31): M = l%16 ; VGPR v: v<4 -> k0 = 2v + (l>=16 ? 8:0),
//                            v>=4 -> k0 = 16 + 2(v-4) + (l>=16 ? 8:0).
// B fragments use the mirrored layout (lane = N column, same K pattern) and
// are pre-packed in LDS as [frag][lane][8 x u32] so each lane loads 32B.
// ---------------------------------------------------------------------------
__device__ __forceinline__ int frag_k0(int v, int lhi) {
    int base = (v < 4) ? (2 * v) : (16 + 2 * (v - 4));
    return base + (lhi ? 8 : 0);
}

__device__ __forceinline__ unsigned int pack2h(float a, float b) {
    _Float16 ha = (_Float16)a, hb = (_Float16)b;
    unsigned short ua = __builtin_bit_cast(unsigned short, ha);
    unsigned short ub = __builtin_bit_cast(unsigned short, hb);
    return ((unsigned int)ub << 16) | (unsigned int)ua;
}

// Build an A fragment (16x32 f16 tile) from a plain row-major f16 buffer.
// rowptr points at element [m][0] of a 64-wide row; kc selects K 0..31/32..63.
__device__ __forceinline__ v16h build_a_frag(const _Float16* rowptr, int kc, int lhi) {
    v8u u;
#pragma unroll
    for (int v = 0; v < 8; ++v) {
        int k0 = frag_k0(v, lhi) + kc * 32;
        u[v] = *(const unsigned int*)(rowptr + k0);  // two adjacent f16 (k0 even)
    }
    return __builtin_bit_cast(v16h, u);
}

__device__ __forceinline__ v8f wmma16(v16h a, v16h b, v8f c) {
    return __builtin_amdgcn_wmma_f32_16x16x32_f16(false, a, false, b, (short)0, c,
                                                  false, false);
}

__device__ __forceinline__ float sigmoidf_(float x) {
    return 1.0f / (1.0f + __expf(-x));
}

// ---------------------------------------------------------------------------
// Encoder: one workgroup per variable n. 256 threads = 8 waves.
// LSTM over T=127 steps; gates[B=32, 4D=256] = h @ Whh^T done with WMMA.
// Online-softmax temporal attention pooling -> c_enc[b*N*D + n*D + d].
// ---------------------------------------------------------------------------
__global__ __launch_bounds__(256)
void enc_kernel(const float* __restrict__ X,        // [B=32, L=128, N=64]
                const float* __restrict__ Wih,      // [N, 256, 1]
                const float* __restrict__ Whh,      // [N, 256, 64]
                const float* __restrict__ bias,     // [N, 256]
                const float* __restrict__ pw,       // [N, 64]
                const float* __restrict__ pb,       // [N]
                float* __restrict__ c_enc)          // [B, N, 64]
{
    const int n = blockIdx.x;
    __shared__ unsigned int wpack[32 * 256];   // 32 frags x 32 lanes x 8 u32 (32 KB)
    __shared__ float gates[32 * 256];          // 32 KB
    __shared__ _Float16 hbuf[32 * 64];         // 4 KB
    __shared__ float biasS[256];
    __shared__ float wihS[256];
    __shared__ float pwS[64];
    __shared__ float part[32][8];
    __shared__ float mS[32], sS[32], alphaS[32], wS[32];

    const int tid = threadIdx.x;
    const float* W = Whh + n * 256 * 64;       // W[j=0..255][k=0..63]

    // ---- pack B fragments: frag = kc*16 + nt; lane holds column nt*16 + l%16
    for (int idx = tid; idx < 32 * 256; idx += 256) {
        int frag = idx >> 8;
        int rem  = idx & 255;
        int lane = rem >> 3;
        int v    = rem & 7;
        int kc   = frag >> 4;
        int nt   = frag & 15;
        int ncol = nt * 16 + (lane & 15);
        int k0   = frag_k0(v, lane >= 16) + kc * 32;
        wpack[frag * 256 + lane * 8 + v] =
            pack2h(W[ncol * 64 + k0], W[ncol * 64 + k0 + 1]);
    }
    if (tid < 256) { biasS[tid] = bias[n * 256 + tid]; wihS[tid] = Wih[n * 256 + tid]; }
    if (tid < 64)  pwS[tid] = pw[n * 64 + tid];
    for (int j = tid; j < 32 * 64; j += 256) hbuf[j] = (_Float16)0.0f;
    if (tid < 32) { mS[tid] = -1e30f; sS[tid] = 0.0f; }
    __syncthreads();

    const int b     = tid >> 3;          // batch row owned in elementwise phase
    const int dbase = (tid & 7) * 8;     // 8 contiguous hidden units
    const int wave  = tid >> 5;
    const int lane  = tid & 31;
    const int lmod  = lane & 15;
    const int lhi   = (lane >= 16);

    float c_reg[8], pool_acc[8], hnew[8];
#pragma unroll
    for (int j = 0; j < 8; ++j) { c_reg[j] = 0.0f; pool_acc[j] = 0.0f; }

    const float pbn = pb[n];

    for (int t = 0; t < 127; ++t) {
        // ---- gates = h @ Whh^T  (WMMA; 8 waves x 4 tiles) ----
        for (int i = 0; i < 4; ++i) {
            int tile = wave + 8 * i;         // 0..31
            int mt = tile >> 4, nt = tile & 15;
            v8f acc = {0.f, 0.f, 0.f, 0.f, 0.f, 0.f, 0.f, 0.f};
#pragma unroll
            for (int kc = 0; kc < 2; ++kc) {
                v16h a = build_a_frag(&hbuf[(mt * 16 + lmod) * 64], kc, lhi);
                v8u bw = *(const v8u*)&wpack[(kc * 16 + nt) * 256 + lane * 8];
                acc = wmma16(a, __builtin_bit_cast(v16h, bw), acc);
            }
#pragma unroll
            for (int r = 0; r < 8; ++r) {
                int m    = mt * 16 + r + (lhi ? 8 : 0);
                int ncol = nt * 16 + lmod;
                gates[m * 256 + ncol] = acc[r];
            }
        }
        __syncthreads();                               // [A]

        // ---- elementwise gates + LSTM update ----
        float xv = X[b * (128 * 64) + t * 64 + n];
        float hp = 0.0f;
#pragma unroll
        for (int j = 0; j < 8; ++j) {
            int d = dbase + j;
            float gi = gates[b * 256 + d]        + biasS[d]        + xv * wihS[d];
            float gf = gates[b * 256 + 64 + d]   + biasS[64 + d]   + xv * wihS[64 + d];
            float gg = gates[b * 256 + 128 + d]  + biasS[128 + d]  + xv * wihS[128 + d];
            float go = gates[b * 256 + 192 + d]  + biasS[192 + d]  + xv * wihS[192 + d];
            float c  = sigmoidf_(gf) * c_reg[j] + sigmoidf_(gi) * tanhf(gg);
            c_reg[j] = c;
            float h  = sigmoidf_(go) * tanhf(c);
            hnew[j]  = h;
            hp += h * pwS[d];
        }
        part[b][tid & 7] = hp;
        __syncthreads();                               // [B]

        if (tid < 32) {
            float lg = pbn;
#pragma unroll
            for (int q = 0; q < 8; ++q) lg += part[tid][q];
            float M  = mS[tid];
            float nm = fmaxf(M, lg);
            float al = __expf(M - nm);
            float w  = __expf(lg - nm);
            sS[tid] = sS[tid] * al + w;
            mS[tid] = nm; alphaS[tid] = al; wS[tid] = w;
        }
        __syncthreads();                               // [C]

        float al = alphaS[b], w = wS[b];
#pragma unroll
        for (int j = 0; j < 8; ++j) {
            pool_acc[j] = pool_acc[j] * al + w * hnew[j];
            hbuf[b * 64 + dbase + j] = (_Float16)hnew[j];
        }
        __syncthreads();                               // [D]
    }

    float inv = 1.0f / sS[b];
#pragma unroll
    for (int j = 0; j < 8; ++j)
        c_enc[b * (64 * 64) + n * 64 + dbase + j] = pool_acc[j] * inv;
}

// ---------------------------------------------------------------------------
// Generic GEMM:  Out[2048][NC] = A[2048][64] @ Wt[NC][64]^T + bias[NC]
// (used for QKV projection NC=192 and output projection NC=64).
// Grid 16 blocks x 128 rows, 8 waves/block, 1 M-tile per wave.
// ---------------------------------------------------------------------------
__global__ __launch_bounds__(256)
void gemm_bias_kernel(const float* __restrict__ A,
                      const float* __restrict__ Wt,
                      const float* __restrict__ bias,
                      float* __restrict__ Out, int NC)
{
    __shared__ _Float16 ain[128 * 64];          // 16 KB
    __shared__ unsigned int wpack[24 * 256];    // up to 24 KB

    const int tid    = threadIdx.x;
    const int ntiles = NC >> 4;
    const int nfrag  = ntiles * 2;
    const int rowbase = blockIdx.x * 128;

    for (int idx = tid; idx < nfrag * 256; idx += 256) {
        int frag = idx >> 8;
        int rem  = idx & 255;
        int lane = rem >> 3;
        int v    = rem & 7;
        int kc   = (frag >= ntiles) ? 1 : 0;
        int nt   = kc ? (frag - ntiles) : frag;
        int ncol = nt * 16 + (lane & 15);
        int k0   = frag_k0(v, lane >= 16) + kc * 32;
        wpack[frag * 256 + lane * 8 + v] =
            pack2h(Wt[ncol * 64 + k0], Wt[ncol * 64 + k0 + 1]);
    }
    for (int idx = tid; idx < 128 * 64; idx += 256)
        ain[idx] = (_Float16)A[rowbase * 64 + idx];
    __syncthreads();

    const int wave = tid >> 5, lane = tid & 31;
    const int lmod = lane & 15, lhi = (lane >= 16);
    const int mt   = wave;

    for (int nt = 0; nt < ntiles; ++nt) {
        v8f acc = {0.f, 0.f, 0.f, 0.f, 0.f, 0.f, 0.f, 0.f};
#pragma unroll
        for (int kc = 0; kc < 2; ++kc) {
            v16h a = build_a_frag(&ain[(mt * 16 + lmod) * 64], kc, lhi);
            v8u bw = *(const v8u*)&wpack[(kc * ntiles + nt) * 256 + lane * 8];
            acc = wmma16(a, __builtin_bit_cast(v16h, bw), acc);
        }
#pragma unroll
        for (int r = 0; r < 8; ++r) {
            int m    = rowbase + mt * 16 + r + (lhi ? 8 : 0);
            int ncol = nt * 16 + lmod;
            Out[m * NC + ncol] = acc[r] + bias[ncol];
        }
    }
}

// ---------------------------------------------------------------------------
// Attention core: one workgroup per (b, head); 64 threads, 1 token each.
// scores over N=64 tokens, DH=16, scale 1/4; softmax; output to osc[B,N,D].
// ---------------------------------------------------------------------------
__global__ __launch_bounds__(64)
void attn_core_kernel(const float* __restrict__ qkv,   // [2048][192] (q|k|v)
                      float* __restrict__ osc)         // [2048][64]
{
    const int bh = blockIdx.x;
    const int b  = bh >> 2, h = bh & 3;
    __shared__ float Ks[64][16];
    __shared__ float Vs[64][16];
    const int i = threadIdx.x;

    const float* base = qkv + (b * 64 + i) * 192 + h * 16;
    float q[16];
#pragma unroll
    for (int d = 0; d < 16; ++d) {
        q[d]     = base[d];
        Ks[i][d] = base[64 + d];
        Vs[i][d] = base[128 + d];
    }
    __syncthreads();

    float mx = -1e30f;
    for (int j = 0; j < 64; ++j) {
        float s = 0.0f;
#pragma unroll
        for (int d = 0; d < 16; ++d) s += q[d] * Ks[j][d];
        mx = fmaxf(mx, s * 0.25f);
    }
    float sum = 0.0f, o[16];
#pragma unroll
    for (int d = 0; d < 16; ++d) o[d] = 0.0f;
    for (int j = 0; j < 64; ++j) {
        float s = 0.0f;
#pragma unroll
        for (int d = 0; d < 16; ++d) s += q[d] * Ks[j][d];
        float e = __expf(s * 0.25f - mx);
        sum += e;
#pragma unroll
        for (int d = 0; d < 16; ++d) o[d] += e * Vs[j][d];
    }
    float inv = 1.0f / sum;
    float* op = osc + (b * 64 + i) * 64 + h * 16;
#pragma unroll
    for (int d = 0; d < 16; ++d) op[d] = o[d] * inv;
}

// ---------------------------------------------------------------------------
// Decoder: one workgroup per variable. Zero-input LSTM, 128 steps.
// h0/c0 = tanh(Cstar @ W^T + b); per-step readout y = h.ow + ob.
// ---------------------------------------------------------------------------
__global__ __launch_bounds__(256)
void dec_kernel(const float* __restrict__ Cstar,   // [B, N, 64] (from d_out)
                const float* __restrict__ Whh,     // [N, 256, 64]
                const float* __restrict__ bias,    // [N, 256]
                const float* __restrict__ WhW, const float* __restrict__ Whb,
                const float* __restrict__ WcW, const float* __restrict__ Wcb,
                const float* __restrict__ ow, const float* __restrict__ ob,
                float* __restrict__ out)           // d_out base
{
    const int n = blockIdx.x;
    __shared__ unsigned int wpack[32 * 256];
    __shared__ float gates[32 * 256];
    __shared__ _Float16 hbuf[32 * 64];
    __shared__ float biasS[256];
    __shared__ float owS[64];
    __shared__ float csS[32 * 64];
    __shared__ float part[32][8];

    const int tid = threadIdx.x;
    const float* W = Whh + n * 256 * 64;

    for (int idx = tid; idx < 32 * 256; idx += 256) {
        int frag = idx >> 8;
        int rem  = idx & 255;
        int lane = rem >> 3;
        int v    = rem & 7;
        int kc   = frag >> 4;
        int nt   = frag & 15;
        int ncol = nt * 16 + (lane & 15);
        int k0   = frag_k0(v, lane >= 16) + kc * 32;
        wpack[frag * 256 + lane * 8 + v] =
            pack2h(W[ncol * 64 + k0], W[ncol * 64 + k0 + 1]);
    }
    if (tid < 256) biasS[tid] = bias[n * 256 + tid];
    if (tid < 64)  owS[tid] = ow[n * 64 + tid];
    for (int idx = tid; idx < 32 * 64; idx += 256) {
        int bb = idx >> 6, k = idx & 63;
        csS[idx] = Cstar[bb * (64 * 64) + n * 64 + k];
    }
    __syncthreads();

    const int b     = tid >> 3;
    const int dbase = (tid & 7) * 8;

    float c_reg[8];
#pragma unroll
    for (int j = 0; j < 8; ++j) {
        int d = dbase + j;
        float ah = Whb[n * 64 + d];
        float ac = Wcb[n * 64 + d];
        const float* wh = WhW + n * 4096 + d * 64;
        const float* wc = WcW + n * 4096 + d * 64;
        for (int k = 0; k < 64; ++k) {
            float cv = csS[b * 64 + k];
            ah += cv * wh[k];
            ac += cv * wc[k];
        }
        c_reg[j] = tanhf(ac);
        hbuf[b * 64 + d] = (_Float16)tanhf(ah);
    }
    __syncthreads();

    const int wave = tid >> 5, lane = tid & 31;
    const int lmod = lane & 15, lhi = (lane >= 16);
    const float obv = ob[n];

    for (int t = 0; t < 128; ++t) {
        for (int i = 0; i < 4; ++i) {
            int tile = wave + 8 * i;
            int mt = tile >> 4, nt = tile & 15;
            v8f acc = {0.f, 0.f, 0.f, 0.f, 0.f, 0.f, 0.f, 0.f};
#pragma unroll
            for (int kc = 0; kc < 2; ++kc) {
                v16h a = build_a_frag(&hbuf[(mt * 16 + lmod) * 64], kc, lhi);
                v8u bw = *(const v8u*)&wpack[(kc * 16 + nt) * 256 + lane * 8];
                acc = wmma16(a, __builtin_bit_cast(v16h, bw), acc);
            }
#pragma unroll
            for (int r = 0; r < 8; ++r) {
                int m    = mt * 16 + r + (lhi ? 8 : 0);
                int ncol = nt * 16 + lmod;
                gates[m * 256 + ncol] = acc[r];
            }
        }
        __syncthreads();                              // gates ready, hbuf reads done

        float hp = 0.0f;
#pragma unroll
        for (int j = 0; j < 8; ++j) {
            int d = dbase + j;
            float gi = gates[b * 256 + d]       + biasS[d];
            float gf = gates[b * 256 + 64 + d]  + biasS[64 + d];
            float gg = gates[b * 256 + 128 + d] + biasS[128 + d];
            float go = gates[b * 256 + 192 + d] + biasS[192 + d];
            float c  = sigmoidf_(gf) * c_reg[j] + sigmoidf_(gi) * tanhf(gg);
            c_reg[j] = c;
            float h  = sigmoidf_(go) * tanhf(c);
            hbuf[b * 64 + d] = (_Float16)h;
            hp += h * owS[d];
        }
        part[b][tid & 7] = hp;
        __syncthreads();

        if (tid < 32) {
            float y = obv;
#pragma unroll
            for (int q = 0; q < 8; ++q) y += part[tid][q];
            if (t < 127) out[tid * (127 * 64) + t * 64 + n] = y;       // recon[b][t][n]
            else         out[260096 + tid * 64 + n] = y;               // pred[b][n]
        }
    }
}

// ---------------------------------------------------------------------------
// Launch
// ---------------------------------------------------------------------------
extern "C" void kernel_launch(void* const* d_in, const int* in_sizes, int n_in,
                              void* d_out, int out_size, void* d_ws, size_t ws_size,
                              hipStream_t stream) {
    const float* X         = (const float*)d_in[0];
    const float* enc_Wih   = (const float*)d_in[1];
    const float* enc_Whh   = (const float*)d_in[2];
    const float* enc_b     = (const float*)d_in[3];
    const float* pool_w    = (const float*)d_in[4];
    const float* pool_b    = (const float*)d_in[5];
    const float* attn_Wqkv = (const float*)d_in[6];
    const float* attn_bqkv = (const float*)d_in[7];
    const float* attn_Wo   = (const float*)d_in[8];
    const float* attn_bo   = (const float*)d_in[9];
    const float* dec_Whh   = (const float*)d_in[11];
    const float* dec_b     = (const float*)d_in[12];
    const float* init_h_W  = (const float*)d_in[13];
    const float* init_h_b  = (const float*)d_in[14];
    const float* init_c_W  = (const float*)d_in[15];
    const float* init_c_b  = (const float*)d_in[16];
    const float* out_w     = (const float*)d_in[17];
    const float* out_b     = (const float*)d_in[18];

    float* out = (float*)d_out;
    float* ws  = (float*)d_ws;
    float* c_enc = ws;                        // 131072 f32  [B,N,D]
    float* qkv   = ws + 131072;               // 393216 f32  [B*N,192]
    float* osc   = ws + 131072 + 393216;      // 131072 f32  [B*N,64]
    float* Cstar = out + 262144;              // third output, also decoder input

    enc_kernel<<<64, 256, 0, stream>>>(X, enc_Wih, enc_Whh, enc_b, pool_w, pool_b, c_enc);
    gemm_bias_kernel<<<16, 256, 0, stream>>>(c_enc, attn_Wqkv, attn_bqkv, qkv, 192);
    attn_core_kernel<<<128, 64, 0, stream>>>(qkv, osc);
    gemm_bias_kernel<<<16, 256, 0, stream>>>(osc, attn_Wo, attn_bo, Cstar, 64);
    dec_kernel<<<64, 256, 0, stream>>>(Cstar, dec_Whh, dec_b,
                                       init_h_W, init_h_b, init_c_W, init_c_b,
                                       out_w, out_b, out);
}
